// Sequence_21277267984819
// MI455X (gfx1250) — matrix operand order repaired and checked by reading, not compiled, not run on hardware
//
#include <hip/hip_runtime.h>
#include <hip/hip_bf16.h>

// ---------------------------------------------------------------------------
// Types for CDNA5 WMMA (wave32): v_wmma_f32_16x16x32_bf16
// ---------------------------------------------------------------------------
typedef __attribute__((ext_vector_type(16))) __bf16       bf16x16;
typedef __attribute__((ext_vector_type(8)))  float        f32x8;
typedef __attribute__((ext_vector_type(4)))  unsigned int u32x4;

union Frag16 { bf16x16 v; u32x4 q[2]; };

__device__ __forceinline__ unsigned short f32_to_bf16(float f) {
  unsigned int u = __float_as_uint(f);
  u += 0x7FFFu + ((u >> 16) & 1u);          // round-to-nearest-even
  return (unsigned short)(u >> 16);
}
__device__ __forceinline__ float sigmoid_f(float x) {
  return 1.0f / (1.0f + __expf(-x));
}
__device__ __forceinline__ float tanh_f(float x) {
  float e = __expf(-2.0f * fabsf(x));
  float r = (1.0f - e) / (1.0f + e);
  return copysignf(r, x);
}
__device__ __forceinline__ f32x8 fzero8() {
  f32x8 z = {0.f, 0.f, 0.f, 0.f, 0.f, 0.f, 0.f, 0.f};
  return z;
}

// A fragment (16x32 bf16, MxK): lane L: m=L%16, g=L/16.
// halves 0..15 <- K = kbase + 16g + j (contiguous, matches packed B)
__device__ __forceinline__ bf16x16 load_a_frag(const unsigned short* rowk, int g) {
  Frag16 f;
  f.q[0] = *(const u32x4*)(rowk + 16 * g);
  f.q[1] = *(const u32x4*)(rowk + 16 * g + 8);
  return f.v;
}
// B fragment pre-packed lane-major: 32 lanes x 16 bf16 contiguous (1KB tile).
__device__ __forceinline__ bf16x16 load_b_frag(const unsigned short* tile, int lane) {
  Frag16 f;
  const u32x4* p = (const u32x4*)tile + lane * 2;
  f.q[0] = p[0];
  f.q[1] = p[1];
  return f.v;
}
__device__ __forceinline__ f32x8 wmma_bf16(bf16x16 a, bf16x16 b, f32x8 c) {
  return __builtin_amdgcn_wmma_f32_16x16x32_bf16(false, a, false, b, (short)0, c,
                                                 false, false);
}

// ---------------------------------------------------------------------------
// Gate accumulation with distance-1 software pipeline: the loads for kt+1
// (A + all four gate B tiles) are issued before the four WMMAs of kt, so the
// L2 latency overlaps one full WMMA group x (4 waves/SIMD).
// ---------------------------------------------------------------------------
template <int KT, int H16, int KC>
__device__ __forceinline__ void cell_gates(const unsigned short* aLDS,
                                           const unsigned short* wpack, int ct,
                                           int lane, f32x8& gi, f32x8& gf,
                                           f32x8& gg, f32x8& go) {
  const int m = lane & 15;
  const int g = lane >> 4;
  const unsigned short* arow = aLDS + m * KC;
  const unsigned short* wi = wpack + (size_t)((0 * H16 + ct) * KT) * 512;
  const unsigned short* wf = wpack + (size_t)((1 * H16 + ct) * KT) * 512;
  const unsigned short* wg = wpack + (size_t)((2 * H16 + ct) * KT) * 512;
  const unsigned short* wo = wpack + (size_t)((3 * H16 + ct) * KT) * 512;

  bf16x16 a  = load_a_frag(arow, g);
  bf16x16 bi = load_b_frag(wi, lane);
  bf16x16 bf = load_b_frag(wf, lane);
  bf16x16 bg = load_b_frag(wg, lane);
  bf16x16 bo = load_b_frag(wo, lane);
  gi = fzero8(); gf = fzero8(); gg = fzero8(); go = fzero8();
#pragma unroll
  for (int kt = 0; kt < KT; ++kt) {
    bf16x16 an, bin, bfn, bgn, bon;
    if (kt + 1 < KT) {                       // prefetch next stage first
      an  = load_a_frag(arow + (kt + 1) * 32, g);
      bin = load_b_frag(wi + (size_t)(kt + 1) * 512, lane);
      bfn = load_b_frag(wf + (size_t)(kt + 1) * 512, lane);
      bgn = load_b_frag(wg + (size_t)(kt + 1) * 512, lane);
      bon = load_b_frag(wo + (size_t)(kt + 1) * 512, lane);
    } else {
      an = a; bin = bi; bfn = bf; bgn = bg; bon = bo;
    }
    gi = wmma_bf16(a, bi, gi);
    gf = wmma_bf16(a, bf, gf);
    gg = wmma_bf16(a, bg, gg);
    go = wmma_bf16(a, bo, go);
    a = an; bi = bin; bf = bfn; bg = bgn; bo = bon;
  }
}

// Elementwise LSTM update (m = r + 8*(lane>=16), n = ct*16 + lane%16).
// Biases are hoisted into registers by the caller.
template <int HX>
__device__ __forceinline__ void lstm_update(const f32x8& gi, const f32x8& gf,
                                            const f32x8& gg, const f32x8& go,
                                            float bi, float bf, float bg,
                                            float bo, int n, int lane,
                                            float* cLDS, unsigned short* h1dst,
                                            int s1, int o1, unsigned short* h2dst,
                                            int s2, int o2) {
  const int mg = (lane >> 4) * 8;
#pragma unroll
  for (int r = 0; r < 8; ++r) {
    const int m = mg + r;
    float iv = sigmoid_f(gi[r] + bi);
    float fv = sigmoid_f(gf[r] + bf);
    float gv = tanh_f(gg[r] + bg);
    float ov = sigmoid_f(go[r] + bo);
    float c = fv * cLDS[m * HX + n] + iv * gv;
    cLDS[m * HX + n] = c;
    unsigned short hb = f32_to_bf16(ov * tanh_f(c));
    h1dst[m * s1 + o1 + n] = hb;
    if (h2dst) h2dst[m * s2 + o2 + n] = hb;
  }
}

// ---------------------------------------------------------------------------
// Pre-pass: pack f32 weights into bf16 WMMA B-fragment tiles.
// dst order: [tile = ntile*KT + kt][lane][half j]; B layout: n = lane%16,
// K-in-tile = j + 16*(lane/16).  Combined K: [W_ih | pad | W_hh].
// ---------------------------------------------------------------------------
__global__ void pack_weights_kernel(const float* __restrict__ Wih,
                                    const float* __restrict__ Whh,
                                    unsigned short* __restrict__ dst, int Kin,
                                    int Ksplit, int Khh, int Kt, int total) {
  int e = blockIdx.x * blockDim.x + threadIdx.x;
  if (e >= total) return;
  int tile = e >> 9;
  int r = e & 511;
  int lane = r >> 4;
  int j = r & 15;
  int ntile = tile / Kt;
  int kt = tile % Kt;
  int N = ntile * 16 + (lane & 15);
  int Kc = kt * 32 + j + 16 * (lane >> 4);
  float v;
  if (Kc < Kin)         v = Wih[(size_t)N * Kin + Kc];
  else if (Kc < Ksplit) v = 0.f;
  else                  v = Whh[(size_t)N * Khh + (Kc - Ksplit)];
  dst[e] = f32_to_bf16(v);
}

// W_lin (2 x 448) -> 14 B tiles, N padded to 16 with zero columns.
__global__ void pack_lin_kernel(const float* __restrict__ W_lin,
                                unsigned short* __restrict__ dst) {
  int e = blockIdx.x * blockDim.x + threadIdx.x;
  if (e >= 14 * 512) return;
  int kt = e >> 9;
  int r = e & 511;
  int lane = r >> 4;
  int j = r & 15;
  int n = lane & 15;
  int Kc = kt * 32 + j + 16 * (lane >> 4);
  float v = (n < 2) ? W_lin[n * 448 + Kc] : 0.f;
  dst[e] = f32_to_bf16(v);
}

__global__ void pack_bias_kernel(const float* bi1, const float* bh1,
                                 const float* bi2, const float* bh2,
                                 const float* bi3, const float* bh3,
                                 float* __restrict__ dst) {
  int i = blockIdx.x * blockDim.x + threadIdx.x;
  if (i < 256)       dst[i] = bi1[i] + bh1[i];
  else if (i < 768)  dst[i] = bi2[i - 256] + bh2[i - 256];
  else if (i < 1792) dst[i] = bi3[i - 768] + bh3[i - 768];
}

// ---------------------------------------------------------------------------
// Persistent recurrence kernel: 64 workgroups x 16 batch rows, 16 waves
// (4 waves/SIMD for latency hiding).  271 sequential steps; state in LDS,
// packed weights streamed from L2.
// ---------------------------------------------------------------------------
__global__ void __launch_bounds__(512, 4) lstm_seq_kernel(
    const float* __restrict__ input, const unsigned short* __restrict__ w1,
    const unsigned short* __restrict__ w2, const unsigned short* __restrict__ w3,
    const unsigned short* __restrict__ wlin, const float* __restrict__ biasws,
    const float* __restrict__ b_lin, float* __restrict__ out) {
  __shared__ __align__(16) unsigned short a1[16 * 96];   // [x(2) pad(30) | h1(64)]
  __shared__ __align__(16) unsigned short a2[16 * 192];  // [h1(64) | h2(128)]
  __shared__ __align__(16) unsigned short a3[16 * 384];  // [h2(128) | h3(256)]
  __shared__ float c1s[16 * 64];
  __shared__ float c2s[16 * 128];
  __shared__ float c3s[16 * 256];

  const int tid = threadIdx.x;
  const int wave = tid >> 5;   // 0..15
  const int lane = tid & 31;
  const int b0 = blockIdx.x * 16;

  for (int i = tid; i < 16 * 96;  i += 512) a1[i] = 0;
  for (int i = tid; i < 16 * 192; i += 512) a2[i] = 0;
  for (int i = tid; i < 16 * 384; i += 512) a3[i] = 0;
  for (int i = tid; i < 16 * 64;  i += 512) c1s[i] = 0.f;
  for (int i = tid; i < 16 * 128; i += 512) c2s[i] = 0.f;
  for (int i = tid; i < 16 * 256; i += 512) c3s[i] = 0.f;
  __syncthreads();

  // Hoist loop-invariant biases into registers.
  const int nl = lane & 15;
  const int n1 = wave * 16 + nl;           // valid when wave < 4
  const int n2 = wave * 16 + nl;           // valid when wave < 8
  const int n3 = wave * 16 + nl;           // valid for all 16 waves
  float b1i = 0.f, b1f = 0.f, b1g = 0.f, b1o = 0.f;
  float b2i = 0.f, b2f = 0.f, b2g = 0.f, b2o = 0.f;
  if (wave < 4) {
    b1i = biasws[n1];       b1f = biasws[64 + n1];
    b1g = biasws[128 + n1]; b1o = biasws[192 + n1];
  }
  if (wave < 8) {
    b2i = biasws[256 + n2];       b2f = biasws[256 + 128 + n2];
    b2g = biasws[256 + 256 + n2]; b2o = biasws[256 + 384 + n2];
  }
  const float b3i = biasws[768 + n3];
  const float b3f = biasws[768 + 256 + n3];
  const float b3g = biasws[768 + 512 + n3];
  const float b3o = biasws[768 + 768 + n3];
  const float blin = (wave == 0 && nl < 2) ? b_lin[nl] : 0.f;

  for (int t = 0; t < 271; ++t) {
    if (wave == 0 && t < 256) {  // sequence phase: x from input
      const int m = lane >> 1, c = lane & 1;
      a1[m * 96 + c] = f32_to_bf16(input[(size_t)(b0 + m) * 512 + t * 2 + c]);
    }
    __syncthreads();  // x + previous projection visible to all waves

    f32x8 gi = fzero8(), gf = fzero8(), gg = fzero8(), go = fzero8();

    // ---- cell 1: H1=64 -> 4 column tiles, waves 0..3 ----
    if (wave < 4) cell_gates<3, 4, 96>(a1, w1, wave, lane, gi, gf, gg, go);
    __syncthreads();
    if (wave < 4)
      lstm_update<64>(gi, gf, gg, go, b1i, b1f, b1g, b1o, n1, lane, c1s, a1, 96,
                      32, a2, 192, 0);
    __syncthreads();

    // ---- cell 2: H2=128 -> 8 column tiles, waves 0..7 ----
    if (wave < 8) cell_gates<6, 8, 192>(a2, w2, wave, lane, gi, gf, gg, go);
    __syncthreads();
    if (wave < 8)
      lstm_update<128>(gi, gf, gg, go, b2i, b2f, b2g, b2o, n2, lane, c2s, a2,
                       192, 64, a3, 384, 0);
    __syncthreads();

    // ---- cell 3: H3=256 -> 16 column tiles, one per wave ----
    cell_gates<12, 16, 384>(a3, w3, wave, lane, gi, gf, gg, go);
    __syncthreads();
    lstm_update<256>(gi, gf, gg, go, b3i, b3f, b3g, b3o, n3, lane, c3s, a3, 384,
                     128, (unsigned short*)nullptr, 0, 0);
    __syncthreads();

    // ---- output projection (448 -> 2) via WMMA, wave 0; feeds next x ----
    if (wave == 0) {
      const int m = lane & 15, g = lane >> 4;
      f32x8 acc = fzero8();
      bf16x16 a = load_a_frag(a2 + m * 192, g);
      bf16x16 b = load_b_frag(wlin, lane);
#pragma unroll
      for (int kt = 0; kt < 14; ++kt) {
        bf16x16 an, bn;
        if (kt + 1 < 14) {
          const unsigned short* src = (kt + 1 < 2)
                                          ? (a2 + m * 192 + (kt + 1) * 32)
                                          : (a3 + m * 384 + (kt - 1) * 32);
          an = load_a_frag(src, g);
          bn = load_b_frag(wlin + (size_t)(kt + 1) * 512, lane);
        } else {
          an = a; bn = b;
        }
        acc = wmma_bf16(a, b, acc);
        a = an; b = bn;
      }
      if (nl < 2) {  // only columns 0,1 are real outputs
#pragma unroll
        for (int r = 0; r < 8; ++r) {
          const int mm = r + 8 * g;
          float v = acc[r] + blin;
          if (t >= 255) out[(size_t)(b0 + mm) * 32 + (t - 255) * 2 + nl] = v;
          a1[mm * 96 + nl] = f32_to_bf16(v);  // next step's x (future phase)
        }
      }
    }
  }
}

// ---------------------------------------------------------------------------
// Launch
// ---------------------------------------------------------------------------
extern "C" void kernel_launch(void* const* d_in, const int* in_sizes, int n_in,
                              void* d_out, int out_size, void* d_ws,
                              size_t ws_size, hipStream_t stream) {
  (void)in_sizes; (void)n_in; (void)out_size; (void)ws_size;
  const float* input = (const float*)d_in[0];
  const float* W_ih1 = (const float*)d_in[2];
  const float* W_hh1 = (const float*)d_in[3];
  const float* b_ih1 = (const float*)d_in[4];
  const float* b_hh1 = (const float*)d_in[5];
  const float* W_ih2 = (const float*)d_in[6];
  const float* W_hh2 = (const float*)d_in[7];
  const float* b_ih2 = (const float*)d_in[8];
  const float* b_hh2 = (const float*)d_in[9];
  const float* W_ih3 = (const float*)d_in[10];
  const float* W_hh3 = (const float*)d_in[11];
  const float* b_ih3 = (const float*)d_in[12];
  const float* b_hh3 = (const float*)d_in[13];
  const float* W_lin = (const float*)d_in[14];
  const float* b_lin = (const float*)d_in[15];

  // Workspace: packed bf16 B-tiles + fused f32 biases (~1.05 MB total).
  unsigned short* w1 = (unsigned short*)d_ws;  // 16nt*3kt  = 48 tiles
  unsigned short* w2 = w1 + 48 * 512;          // 32nt*6kt  = 192 tiles
  unsigned short* w3 = w2 + 192 * 512;         // 64nt*12kt = 768 tiles
  float* biasws = (float*)(w3 + 768 * 512);    // 1792 f32
  unsigned short* wlin = (unsigned short*)(biasws + 1792);  // 14 tiles

  pack_weights_kernel<<<(48 * 512) / 256, 256, 0, stream>>>(
      W_ih1, W_hh1, w1, 2, 32, 64, 3, 48 * 512);
  pack_weights_kernel<<<(192 * 512) / 256, 256, 0, stream>>>(
      W_ih2, W_hh2, w2, 64, 64, 128, 6, 192 * 512);
  pack_weights_kernel<<<(768 * 512) / 256, 256, 0, stream>>>(
      W_ih3, W_hh3, w3, 128, 128, 256, 12, 768 * 512);
  pack_lin_kernel<<<28, 256, 0, stream>>>(W_lin, wlin);
  pack_bias_kernel<<<7, 256, 0, stream>>>(b_ih1, b_hh1, b_ih2, b_hh2, b_ih3,
                                          b_hh3, biasws);

  lstm_seq_kernel<<<64, 512, 0, stream>>>(input, w1, w2, w3, wlin, biasws,
                                          b_lin, (float*)d_out);
}
